// VectorQuantizer2_18897856102744
// MI455X (gfx1250) — matrix operand order repaired and checked by reading, not compiled, not run on hardware
//
#include <hip/hip_runtime.h>
#include <stdint.h>

// Multi-scale residual VQ for MI455X (gfx1250, wave32).
// Core: fused Z@E^T + argmin using v_wmma_f32_16x16x32_f16 (K=32 == C),
// LDS-staged f16 codebook chunks (bank-conflict-padded), branch-free
// packed-key argmin (v_fma / v_and_or / v_min_u32 per slot), bicubic
// upsample with fused embedding gather.

typedef __attribute__((ext_vector_type(16))) _Float16 v16h;
typedef __attribute__((ext_vector_type(8)))  float    v8f;

#define B_      64
#define C_      32
#define H_      16
#define W_      16
#define VOCAB   4096
#define CHUNK   512   // codes staged in LDS per iteration
#define WPB     8     // waves per block in vq kernel (256 threads)
#define LSTRIDE 40    // halfs per code row in LDS (80B: 16B-aligned, bank-spread)
#define DBIAS   0.125f

// ---- init: zero d_out (BCHW), transpose f BCHW -> f_rest BHWC ----
__global__ __launch_bounds__(256) void init_kernel(const float* __restrict__ f,
                                                   float* __restrict__ out,
                                                   float* __restrict__ frest) {
  int i = blockIdx.x * 256 + threadIdx.x;          // BCHW flat index
  out[i] = 0.0f;
  int w = i & 15, h = (i >> 4) & 15, c = (i >> 8) & 31, b = i >> 13;
  frest[((b * H_ + h) * W_ + w) * C_ + c] = f[i];
}

// ---- codebook prep: f32 -> f16 copy + ||e||^2 in f32 ----
__global__ __launch_bounds__(256) void prep_emb_kernel(const float* __restrict__ emb,
                                                       _Float16* __restrict__ emb16,
                                                       float* __restrict__ esq) {
  int j = blockIdx.x * 256 + threadIdx.x;
  float s = 0.f;
#pragma unroll
  for (int c = 0; c < C_; ++c) {
    float v = emb[j * C_ + c];
    s += v * v;
    emb16[j * C_ + c] = (_Float16)v;
  }
  esq[j] = s;
}

// ---- area pooling (torch adaptive_avg_pool windows); writes z as f16 ----
__global__ __launch_bounds__(256) void pool_kernel(const float* __restrict__ frest,
                                                   _Float16* __restrict__ z16,
                                                   int pn, int total) {
  int i = blockIdx.x * 256 + threadIdx.x;
  if (i >= total) return;
  int c = i & 31;
  int cell = i >> 5;
  int q = cell % pn;
  int t = cell / pn;
  int p = t % pn;
  int b = t / pn;
  int sy = (p * H_) / pn, ey = ((p + 1) * H_ + pn - 1) / pn;   // ceil
  int sx = (q * W_) / pn, ex = ((q + 1) * W_ + pn - 1) / pn;
  float s = 0.f;
  for (int y = sy; y < ey; ++y)
    for (int x = sx; x < ex; ++x)
      s += frest[((b * H_ + y) * W_ + x) * C_ + c];
  s *= 1.0f / (float)((ey - sy) * (ex - sx));
  z16[i] = (_Float16)s;
}

// ---- fused GEMM(Z@E^T) + argmin over 4096 codes via WMMA f16 ----
// One wave = one 16-row tile. A (16x32 f16) lives in 8 VGPRs for the whole
// codebook sweep. Distance key: biased-positive float bits with the 12-bit
// code packed into the mantissa LSBs -> argmin is a single v_min_u32 chain.
__global__ __launch_bounds__(256) void vq_kernel(const _Float16* __restrict__ z16,
                                                 const _Float16* __restrict__ emb16,
                                                 const float* __restrict__ esq,
                                                 int* __restrict__ idx, int nrows) {
  __shared__ _Float16 lemb[CHUNK * LSTRIDE];   // 40 KB
  __shared__ float    lesq[CHUNK];             // 2 KB  (holds ||e||^2 + DBIAS)
  const int lane = threadIdx.x & 31;
  const int wave = threadIdx.x >> 5;
  const int tile = blockIdx.x * WPB + wave;
  const int ntiles = nrows >> 4;
  const bool active = tile < ntiles;
  const int row0 = active ? tile * 16 : 0;   // clamp for padding waves (keep EXEC full)
  const int col = lane & 15;
  const int hi  = lane >> 4;

  // A k order used here (contiguous, matched by B; not the ISA table order): lanes0-15 row=lane hold K0..15;
  // lanes16-31 same rows hold K16..31.
  v16h a;
  {
    const _Float16* zr = z16 + (size_t)(row0 + col) * C_;
#pragma unroll
    for (int e = 0; e < 8; ++e) a[e]     = zr[16 * hi + e];
#pragma unroll
    for (int e = 0; e < 8; ++e) a[8 + e] = zr[16 * hi + 8 + e];
  }

  uint32_t bk[8];
#pragma unroll
  for (int v = 0; v < 8; ++v) bk[v] = 0xFFFFFFFFu;

  for (int ch = 0; ch < VOCAB; ch += CHUNK) {
    { // cooperative LDS staging with row padding (uniform trips -> no divergence)
      const uint32_t* src = (const uint32_t*)(emb16 + (size_t)ch * C_);
      uint32_t* dst = (uint32_t*)lemb;
      for (int k = threadIdx.x; k < CHUNK * (C_ / 2); k += 256) {
        int j = k >> 4, d = k & 15;                // code j, dword d
        dst[j * (LSTRIDE / 2) + d] = src[k];
      }
      for (int k = threadIdx.x; k < CHUNK; k += 256) lesq[k] = esq[ch + k] + DBIAS;
    }
    __syncthreads();
#pragma unroll 2
    for (int ct = 0; ct < CHUNK / 16; ++ct) {
      // B k order used here (contiguous, same as A): lanes0-15 col=lane hold K0..15, lanes16-31 K16..31.
      v16h bm;
      const _Float16* er = lemb + ((ct * 16 + col) * LSTRIDE + 16 * hi);
#pragma unroll
      for (int e = 0; e < 16; ++e) bm[e] = er[e];
      v8f acc = {0.f, 0.f, 0.f, 0.f, 0.f, 0.f, 0.f, 0.f};
      acc = __builtin_amdgcn_wmma_f32_16x16x32_f16(
          /*neg_a=*/false, a, /*neg_b=*/false, bm,
          /*c_mod=*/(short)0, acc, /*reuse_a=*/false, /*reuse_b=*/false);
      const float    eqb  = lesq[ct * 16 + col];              // ||e||^2 + bias
      const uint32_t code = (uint32_t)(ch + ct * 16 + col);   // < 4096
      // D layout: lane holds D[v + 8*hi][col] in acc[v]
#pragma unroll
      for (int v = 0; v < 8; ++v) {
        float dist = fmaf(-2.0f, acc[v], eqb);                // positive by bias
        uint32_t key = (__float_as_uint(dist) & 0xFFFFF000u) | code;
        bk[v] = key < bk[v] ? key : bk[v];                    // v_min_u32
      }
    }
    __syncthreads();
  }

  // min across the 16 lanes of each half (xor masks stay in-half)
#pragma unroll
  for (int v = 0; v < 8; ++v) {
    uint32_t k = bk[v];
#pragma unroll
    for (int m = 1; m < 16; m <<= 1) {
      uint32_t ok = (uint32_t)__shfl_xor((int)k, m, 32);
      k = ok < k ? ok : k;
    }
    if (active && col == 0) idx[row0 + v + 8 * hi] = (int)(k & 0xFFFu);
  }
}

// ---- Keys cubic (a = -0.5), jax.image.resize-style renormalized weights ----
__device__ __forceinline__ float keys_cubic(float x) {
  float a = fabsf(x);
  if (a <= 1.0f) return (1.5f * a - 2.5f) * a * a + 1.0f;
  if (a < 2.0f)  return ((-0.5f * a + 2.5f) * a - 4.0f) * a + 2.0f;
  return 0.0f;
}

// ---- bicubic upsample of emb[idx] to 16x16, fused gather + accumulate ----
// out (BCHW) += h_up ; frest (BHWC) -= h_up.  pn==16 degenerates to identity.
__global__ __launch_bounds__(256) void up_kernel(const int* __restrict__ idx,
                                                 const float* __restrict__ emb,
                                                 float* __restrict__ out,
                                                 float* __restrict__ frest,
                                                 int pn) {
  int i = blockIdx.x * 256 + threadIdx.x;  // BHWC flat index
  int c = i & 31, x = (i >> 5) & 15, y = (i >> 9) & 15, b = i >> 13;

  float wy[4], wx[4];
  int   jy[4], jx[4];
  {
    float sf = (y + 0.5f) * (float)pn * (1.0f / 16.0f) - 0.5f;
    int fl = (int)floorf(sf);
    float ws = 0.f;
#pragma unroll
    for (int t = 0; t < 4; ++t) {
      int j = fl - 1 + t;
      float w = (j >= 0 && j < pn) ? keys_cubic(sf - (float)j) : 0.0f;
      jy[t] = j < 0 ? 0 : (j >= pn ? pn - 1 : j);
      wy[t] = w; ws += w;
    }
    float inv = 1.0f / ws;
#pragma unroll
    for (int t = 0; t < 4; ++t) wy[t] *= inv;
  }
  {
    float sf = (x + 0.5f) * (float)pn * (1.0f / 16.0f) - 0.5f;
    int fl = (int)floorf(sf);
    float ws = 0.f;
#pragma unroll
    for (int t = 0; t < 4; ++t) {
      int j = fl - 1 + t;
      float w = (j >= 0 && j < pn) ? keys_cubic(sf - (float)j) : 0.0f;
      jx[t] = j < 0 ? 0 : (j >= pn ? pn - 1 : j);
      wx[t] = w; ws += w;
    }
    float inv = 1.0f / ws;
#pragma unroll
    for (int t = 0; t < 4; ++t) wx[t] *= inv;
  }

  float h = 0.f;
  const int* ib = idx + b * pn * pn;
#pragma unroll
  for (int ty = 0; ty < 4; ++ty) {
    float wyt = wy[ty];
    const int* ir = ib + jy[ty] * pn;
#pragma unroll
    for (int tx = 0; tx < 4; ++tx)
      h += wyt * wx[tx] * emb[ir[jx[tx]] * C_ + c];  // idx uniform per wave (c fastest)
  }
  out[((b * C_ + c) * H_ + y) * W_ + x] += h;
  frest[i] -= h;
}

extern "C" void kernel_launch(void* const* d_in, const int* in_sizes, int n_in,
                              void* d_out, int out_size, void* d_ws, size_t ws_size,
                              hipStream_t stream) {
  const float* f   = (const float*)d_in[0];   // [64,32,16,16] f32
  const float* emb = (const float*)d_in[1];   // [4096,32] f32
  float* out = (float*)d_out;                 // [64,32,16,16] f32

  // workspace carve-up (~3.4 MB total)
  char* ws = (char*)d_ws;
  float* frest = (float*)ws;      ws += (size_t)B_ * H_ * W_ * C_ * sizeof(float);   // 2 MB
  float* esq   = (float*)ws;      ws += (size_t)VOCAB * sizeof(float);               // 16 KB
  _Float16* emb16 = (_Float16*)ws; ws += (size_t)VOCAB * C_ * sizeof(_Float16);      // 256 KB
  _Float16* z16   = (_Float16*)ws; ws += (size_t)B_ * 256 * C_ * sizeof(_Float16);   // 1 MB
  int* idx = (int*)ws;                                                               // 64 KB

  const int NTOT = B_ * C_ * H_ * W_;  // 524288 (multiple of 256)
  init_kernel<<<NTOT / 256, 256, 0, stream>>>(f, out, frest);
  prep_emb_kernel<<<VOCAB / 256, 256, 0, stream>>>(emb, emb16, esq);

  const int pns[10] = {1, 2, 3, 4, 5, 6, 8, 10, 13, 16};
  for (int si = 0; si < 10; ++si) {
    int pn = pns[si];
    int nrows = B_ * pn * pn;           // always a multiple of 16
    int ztot = nrows * C_;              // always a multiple of 256
    pool_kernel<<<(ztot + 255) / 256, 256, 0, stream>>>(frest, z16, pn, ztot);
    int ntiles = nrows / 16;
    int blocks = (ntiles + WPB - 1) / WPB;
    vq_kernel<<<blocks, 256, 0, stream>>>(z16, emb16, esq, idx, nrows);
    up_kernel<<<NTOT / 256, 256, 0, stream>>>(idx, emb, out, frest, pn);
  }
}
